// Multihead_Attention_11708080849464
// MI455X (gfx1250) — compile-verified, matrix operand order checked by reading, not run on hardware
//
#include <hip/hip_runtime.h>

// ---------------------------------------------------------------------------
// MHA forward for MI455X (gfx1250): bf16 WMMA (v_wmma_f32_16x16x32_bf16)
// everywhere, f32 accumulate. One-time packed f32->bf16 conversion of x and
// all weights so every GEMM/attention operand load is a 16B bf16 load.
// Flash-style attention with LDS staging + next-tile global prefetch.
// ---------------------------------------------------------------------------

typedef __attribute__((ext_vector_type(16))) __bf16 v16bf;
typedef __attribute__((ext_vector_type(8)))  __bf16 v8bf;
typedef __attribute__((ext_vector_type(8)))  float  v8f;

#define DIM     1024
#define HEADS   16
#define HD      64
#define SEQ     2048
#define BATCH   2
#define ROWS    (BATCH * SEQ)          // 4096
#define L2E     1.44269504088896340736f
#define SCALE   0.125f                 // 1/sqrt(HD)

static __device__ __forceinline__ v8f v8f_zero() {
  v8f z = {0.f, 0.f, 0.f, 0.f, 0.f, 0.f, 0.f, 0.f};
  return z;
}

// Fragment loader from a row-major bf16 row. Per ISA 7.12.2 (16-bit A 16x32):
// lanes 0-15 hold K in {kb..kb+7} U {16+kb..16+kb+7} with kb=0, lanes 16-31
// with kb=8. Two 16B loads.
static __device__ __forceinline__ v16bf load_frag(const __bf16* rowk, int kb) {
  union { v16bf v; float4 q[2]; } u;
  u.q[0] = *reinterpret_cast<const float4*>(rowk + kb);
  u.q[1] = *reinterpret_cast<const float4*>(rowk + 16 + kb);
  return u.v;
}

static __device__ __forceinline__ v8f wmma_bf16(v16bf a, v16bf b, v8f c) {
  // (neg_a, A, neg_b, B, c_mod, C, reuse_a, reuse_b)
  return __builtin_amdgcn_wmma_f32_16x16x32_bf16(false, a, false, b,
                                                 (short)0, c, false, false);
}

static __device__ __forceinline__ float red_max16(float v) {
  v = fmaxf(v, __shfl_xor(v, 1, 32));
  v = fmaxf(v, __shfl_xor(v, 2, 32));
  v = fmaxf(v, __shfl_xor(v, 4, 32));
  v = fmaxf(v, __shfl_xor(v, 8, 32));
  return v;
}
static __device__ __forceinline__ float red_sum16(float v) {
  v += __shfl_xor(v, 1, 32);
  v += __shfl_xor(v, 2, 32);
  v += __shfl_xor(v, 4, 32);
  v += __shfl_xor(v, 8, 32);
  return v;
}

// ---------------------------------------------------------------------------
// Packed f32 -> bf16 conversion: 8 elements per thread, 16B vector store
// (lowers to v_cvt_pk_bf16_f32 pairs + global_store_b128).
// ---------------------------------------------------------------------------
__global__ __launch_bounds__(256) void cvt_bf16_kernel(
    const float* __restrict__ src, __bf16* __restrict__ dst, int n8) {
  const int i = blockIdx.x * 256 + threadIdx.x;
  if (i >= n8) return;
  const float4* s4 = reinterpret_cast<const float4*>(src) + 2 * (size_t)i;
  float4 a = s4[0], b = s4[1];
  union { v8bf v; float4 q; } u;
  u.v[0] = (__bf16)a.x; u.v[1] = (__bf16)a.y;
  u.v[2] = (__bf16)a.z; u.v[3] = (__bf16)a.w;
  u.v[4] = (__bf16)b.x; u.v[5] = (__bf16)b.y;
  u.v[6] = (__bf16)b.z; u.v[7] = (__bf16)b.w;
  *(reinterpret_cast<float4*>(dst) + i) = u.q;
}

// ---------------------------------------------------------------------------
// GEMM: Y[M,N] = A[M,K] @ W[N,K]^T + bias[N], all operands bf16, f32 accum.
// Each wave owns a 16x64 tile (A-frag reused across 4 B-frags).
// Block = 8 waves -> 64x128 block tile.
// ---------------------------------------------------------------------------
template <bool OUT_BF16>
__global__ __launch_bounds__(256) void gemm_bias_kernel(
    const __bf16* __restrict__ A, const __bf16* __restrict__ W,
    const float* __restrict__ bias, void* __restrict__ outv,
    int M, int N, int K) {
  const int tid   = threadIdx.x;
  const int lane  = tid & 31;
  const int w     = tid >> 5;        // 0..7
  const int wm    = w & 3, wn = w >> 2;
  const int mtile = blockIdx.x * 64 + wm * 16;
  const int ntile = blockIdx.y * 128 + wn * 64;
  const int lm    = lane & 15;
  const int kb    = (lane >> 4) * 8;
  const int roff  = (lane >> 4) * 8;

  v8f acc[4];
#pragma unroll
  for (int c = 0; c < 4; ++c) acc[c] = v8f_zero();

  const __bf16* arow = A + (size_t)(mtile + lm) * K;
  const __bf16* wrow0 = W + (size_t)(ntile + lm) * K;
  for (int k0 = 0; k0 < K; k0 += 32) {
    v16bf af = load_frag(arow + k0, kb);
#pragma unroll
    for (int c = 0; c < 4; ++c) {
      v16bf bf = load_frag(wrow0 + (size_t)(c * 16) * K + k0, kb);
      acc[c] = wmma_bf16(af, bf, acc[c]);
    }
  }

#pragma unroll
  for (int c = 0; c < 4; ++c) {
    const int col = ntile + c * 16 + lm;
    const float bv = bias[col];
#pragma unroll
    for (int r = 0; r < 8; ++r) {
      const int row = mtile + roff + r;   // C-layout: VGPR r -> M = r (+8 hi half)
      const float val = acc[c][r] + bv;
      if (OUT_BF16)
        ((__bf16*)outv)[(size_t)row * N + col] = (__bf16)val;
      else
        ((float*)outv)[(size_t)row * N + col] = val;
    }
  }
}

// ---------------------------------------------------------------------------
// Flash attention forward: one wave per (batch, head, 16-query tile).
// Online softmax over 32-key tiles; P and V staged through LDS so the
// P@V B-fragment gather stays on-chip. Emits O (bf16) and row stats m,l.
// ---------------------------------------------------------------------------
__global__ __launch_bounds__(128) void flash_fwd_kernel(
    const __bf16* __restrict__ Qb, const __bf16* __restrict__ Kb,
    const __bf16* __restrict__ Vb, __bf16* __restrict__ Ob,
    float* __restrict__ Mst, float* __restrict__ Lst) {
  __shared__ __bf16 pbuf[4][16][36];
  __shared__ __align__(16) __bf16 vbuf[4][32][72];

  const int tid  = threadIdx.x;
  const int lane = tid & 31;
  const int w    = tid >> 5;                 // wave in block (4)
  const int task = blockIdx.x * 4 + w;       // 4096 tasks
  const int qt   = task & 127;
  const int h    = (task >> 7) & 15;
  const int b    = task >> 11;
  const int lm   = lane & 15;
  const int kb   = (lane >> 4) * 8;
  const int roff = (lane >> 4) * 8;
  const size_t baserow = (size_t)b * SEQ;
  const int qbase = qt * 16;

  // Q A-fragments, d-steps 0..31 / 32..63
  const __bf16* qrow = Qb + (baserow + qbase + lm) * DIM + h * HD;
  const v16bf aq0 = load_frag(qrow, kb);
  const v16bf aq1 = load_frag(qrow + 32, kb);

  float m_run[8], l_run[8];
#pragma unroll
  for (int r = 0; r < 8; ++r) { m_run[r] = -1e30f; l_run[r] = 0.f; }
  v8f o[4];
#pragma unroll
  for (int c = 0; c < 4; ++c) o[c] = v8f_zero();

  for (int kt = 0; kt < SEQ; kt += 32) {
    // prefetch next key/value tiles into cache (global_prefetch_b8)
    if (kt + 32 < SEQ) {
      __builtin_prefetch(Kb + (baserow + kt + 32 + lane) * DIM + h * HD, 0, 1);
      __builtin_prefetch(Vb + (baserow + kt + 32 + lane) * DIM + h * HD, 0, 1);
    }

    // ---- scores S = Q K^T * scale (two 16-key chunks) ----
    v8f s[2];
#pragma unroll
    for (int c = 0; c < 2; ++c) {
      const __bf16* krow = Kb + (baserow + kt + c * 16 + lm) * DIM + h * HD;
      v8f t = v8f_zero();
      t = wmma_bf16(aq0, load_frag(krow, kb), t);
      t = wmma_bf16(aq1, load_frag(krow + 32, kb), t);
#pragma unroll
      for (int r = 0; r < 8; ++r) t[r] *= SCALE;
      s[c] = t;
    }

    // ---- online softmax update ----
    float mnew[8], alpha[8];
#pragma unroll
    for (int r = 0; r < 8; ++r) {
      float loc = red_max16(fmaxf(s[0][r], s[1][r]));
      mnew[r]  = fmaxf(m_run[r], loc);
      alpha[r] = exp2f((m_run[r] - mnew[r]) * L2E);
      m_run[r] = mnew[r];
    }
#pragma unroll
    for (int r = 0; r < 8; ++r) {
      const float p0 = exp2f((s[0][r] - mnew[r]) * L2E);
      const float p1 = exp2f((s[1][r] - mnew[r]) * L2E);
      s[0][r] = p0; s[1][r] = p1;
      l_run[r] = l_run[r] * alpha[r] + red_sum16(p0 + p1);
#pragma unroll
      for (int c = 0; c < 4; ++c) o[c][r] *= alpha[r];
    }

    __syncthreads();  // previous iteration's LDS reads done (uniform loop)
    // stage P (C-layout -> LDS) and the 32x64 V tile
#pragma unroll
    for (int c = 0; c < 2; ++c)
#pragma unroll
      for (int r = 0; r < 8; ++r)
        pbuf[w][roff + r][c * 16 + lm] = (__bf16)s[c][r];

    const float4* vsrc =
        reinterpret_cast<const float4*>(Vb + (baserow + kt + lane) * DIM + h * HD);
    float4* vdst = reinterpret_cast<float4*>(&vbuf[w][lane][0]);
#pragma unroll
    for (int j = 0; j < 8; ++j) vdst[j] = vsrc[j];
    __syncthreads();

    // P as A-fragment (K-dim = 32 keys), V as B-fragments from LDS
    v16bf ap;
#pragma unroll
    for (int j = 0; j < 8; ++j) {
      ap[j]     = pbuf[w][lm][kb + j];
      ap[8 + j] = pbuf[w][lm][16 + kb + j];
    }
#pragma unroll
    for (int c = 0; c < 4; ++c) {
      v16bf bv;
#pragma unroll
      for (int j = 0; j < 8; ++j) {
        bv[j]     = vbuf[w][kb + j][c * 16 + lm];
        bv[8 + j] = vbuf[w][16 + kb + j][c * 16 + lm];
      }
      o[c] = wmma_bf16(ap, bv, o[c]);
    }
  }

  // ---- epilogue: normalize, store O and stats ----
  float inv[8];
#pragma unroll
  for (int r = 0; r < 8; ++r) inv[r] = 1.f / l_run[r];
#pragma unroll
  for (int c = 0; c < 4; ++c)
#pragma unroll
    for (int r = 0; r < 8; ++r) {
      const int row = qbase + roff + r;
      const int col = h * HD + c * 16 + lm;
      Ob[(baserow + row) * DIM + col] = (__bf16)(o[c][r] * inv[r]);
    }
  if (lm == 0) {
#pragma unroll
    for (int r = 0; r < 8; ++r) {
      const size_t idx = (size_t)(b * HEADS + h) * SEQ + qbase + roff + r;
      Mst[idx] = m_run[r];
      Lst[idx] = l_run[r];
    }
  }
}

// ---------------------------------------------------------------------------
// Head-mean of attention probabilities: recompute scores from the same bf16
// Q/K (bit-identical WMMA), apply stored (m,l), average over heads in regs.
// One wave per (b, 16-query tile, 32-key tile). Deterministic, no atomics.
// ---------------------------------------------------------------------------
__global__ __launch_bounds__(256) void mean_probs_kernel(
    const __bf16* __restrict__ Qb, const __bf16* __restrict__ Kb,
    const float* __restrict__ Mst, const float* __restrict__ Lst,
    float* __restrict__ out_mean) {
  const int tid  = threadIdx.x;
  const int lane = tid & 31;
  const int w    = tid >> 5;
  const int task = blockIdx.x * 8 + w;        // 16384 tasks
  const int kt   = (task & 63) * 32;
  const int qt   = ((task >> 6) & 127) * 16;
  const int b    = task >> 13;
  const int lm   = lane & 15;
  const int kb   = (lane >> 4) * 8;
  const int roff = (lane >> 4) * 8;

  v8f acc[2];
  acc[0] = v8f_zero(); acc[1] = v8f_zero();

  for (int h = 0; h < HEADS; ++h) {
    const __bf16* qrow = Qb + ((size_t)b * SEQ + qt + lm) * DIM + h * HD;
    const v16bf aq0 = load_frag(qrow, kb);
    const v16bf aq1 = load_frag(qrow + 32, kb);

    float mr[8], il[8];
    const size_t sbase = (size_t)(b * HEADS + h) * SEQ + qt + roff;
#pragma unroll
    for (int r = 0; r < 8; ++r) {
      mr[r] = Mst[sbase + r];
      il[r] = (1.f / (float)HEADS) / Lst[sbase + r];
    }
#pragma unroll
    for (int c = 0; c < 2; ++c) {
      const __bf16* krow = Kb + ((size_t)b * SEQ + kt + c * 16 + lm) * DIM + h * HD;
      v8f t = v8f_zero();
      t = wmma_bf16(aq0, load_frag(krow, kb), t);
      t = wmma_bf16(aq1, load_frag(krow + 32, kb), t);
#pragma unroll
      for (int r = 0; r < 8; ++r)
        acc[c][r] += exp2f((t[r] * SCALE - mr[r]) * L2E) * il[r];
    }
  }

#pragma unroll
  for (int c = 0; c < 2; ++c)
#pragma unroll
    for (int r = 0; r < 8; ++r) {
      const int q = qt + roff + r;
      const int k = kt + c * 16 + lm;
      out_mean[((size_t)b * SEQ + q) * SEQ + k] = acc[c][r];
    }
}

// ---------------------------------------------------------------------------
extern "C" void kernel_launch(void* const* d_in, const int* in_sizes, int n_in,
                              void* d_out, int out_size, void* d_ws, size_t ws_size,
                              hipStream_t stream) {
  const float* x  = (const float*)d_in[0];
  const float* wq = (const float*)d_in[1];
  const float* bq = (const float*)d_in[2];
  const float* wk = (const float*)d_in[3];
  const float* bk = (const float*)d_in[4];
  const float* wv = (const float*)d_in[5];
  const float* bv = (const float*)d_in[6];
  const float* wo = (const float*)d_in[7];
  const float* bo = (const float*)d_in[8];

  float* out      = (float*)d_out;                       // [B,S,D] f32
  float* out_mean = out + (size_t)BATCH * SEQ * DIM;     // [B,S,S] f32

  char* ws = (char*)d_ws;
  __bf16* Qb  = (__bf16*)(ws);                           // 8 MiB
  __bf16* Kb  = (__bf16*)(ws + ((size_t)8  << 20));      // 8 MiB
  __bf16* Vb  = (__bf16*)(ws + ((size_t)16 << 20));      // 8 MiB
  __bf16* Ob  = (__bf16*)(ws + ((size_t)24 << 20));      // 8 MiB
  __bf16* Xb  = (__bf16*)(ws + ((size_t)32 << 20));      // 8 MiB
  __bf16* Wqb = (__bf16*)(ws + ((size_t)40 << 20));      // 2 MiB
  __bf16* Wkb = (__bf16*)(ws + ((size_t)42 << 20));      // 2 MiB
  __bf16* Wvb = (__bf16*)(ws + ((size_t)44 << 20));      // 2 MiB
  __bf16* Wob = (__bf16*)(ws + ((size_t)46 << 20));      // 2 MiB
  float*  Mst = (float*)(ws + ((size_t)48 << 20));       // 256 KiB
  float*  Lst = Mst + (size_t)BATCH * HEADS * SEQ;       // 256 KiB

  // ---- one-time packed f32 -> bf16 conversions ----
  const int nx8 = (ROWS * DIM) / 8;      // 524288
  const int nw8 = (DIM * DIM) / 8;       // 131072
  cvt_bf16_kernel<<<(nx8 + 255) / 256, 256, 0, stream>>>(x,  Xb,  nx8);
  cvt_bf16_kernel<<<(nw8 + 255) / 256, 256, 0, stream>>>(wq, Wqb, nw8);
  cvt_bf16_kernel<<<(nw8 + 255) / 256, 256, 0, stream>>>(wk, Wkb, nw8);
  cvt_bf16_kernel<<<(nw8 + 255) / 256, 256, 0, stream>>>(wv, Wvb, nw8);
  cvt_bf16_kernel<<<(nw8 + 255) / 256, 256, 0, stream>>>(wo, Wob, nw8);

  const dim3 gemm_grid(ROWS / 64, DIM / 128);            // 64 x 8 blocks

  // Q/K/V projections (bf16 in, bf16 out)
  gemm_bias_kernel<true><<<gemm_grid, 256, 0, stream>>>(Xb, Wqb, bq, Qb, ROWS, DIM, DIM);
  gemm_bias_kernel<true><<<gemm_grid, 256, 0, stream>>>(Xb, Wkb, bk, Kb, ROWS, DIM, DIM);
  gemm_bias_kernel<true><<<gemm_grid, 256, 0, stream>>>(Xb, Wvb, bv, Vb, ROWS, DIM, DIM);

  // Flash attention (O bf16 + row stats)
  flash_fwd_kernel<<<1024, 128, 0, stream>>>(Qb, Kb, Vb, Ob, Mst, Lst);

  // Output projection (bf16 in, f32 out)
  gemm_bias_kernel<false><<<gemm_grid, 256, 0, stream>>>(Ob, Wob, bo, out, ROWS, DIM, DIM);

  // Head-mean attention probabilities
  mean_probs_kernel<<<2048, 256, 0, stream>>>(Qb, Kb, Mst, Lst, out_mean);
}
